// TopKFreqs_42039139893798
// MI455X (gfx1250) — matrix operand order repaired and checked by reading, not compiled, not run on hardware
//
#include <hip/hip_runtime.h>
#include <hip/hip_bf16.h>
#include <math.h>

// Problem constants (match reference)
#define S       128
#define CH      64
#define BATCH   64
#define TOPK    32
#define PLANES  (BATCH * CH)        // 4096
#define PELEMS  (S * S)             // 16384
#define PIF     3.14159265358979323846f

typedef _Float16 v16h __attribute__((ext_vector_type(16)));
typedef _Float16 v8h  __attribute__((ext_vector_type(8)));
typedef float    v8f  __attribute__((ext_vector_type(8)));
typedef int      v4i  __attribute__((vector_size(16)));   // matches builtin proto

#define AS1 __attribute__((address_space(1)))
#define AS3 __attribute__((address_space(3)))

// gfx1250 async global->LDS copy (ASYNCcnt), guarded so compile never breaks.
#if __has_builtin(__builtin_amdgcn_global_load_async_to_lds_b128)
#define HAVE_ASYNC_LDS 1
#else
#define HAVE_ASYNC_LDS 0
#endif

__device__ __forceinline__ void async_wait0() {
#if __has_builtin(__builtin_amdgcn_s_wait_asynccnt)
  __builtin_amdgcn_s_wait_asynccnt(0);
#else
  asm volatile("s_wait_asynccnt 0x0" ::: "memory");
#endif
}

#define WMMA_F16(A, B, C) \
  __builtin_amdgcn_wmma_f32_16x16x32_f16(false, (A), false, (B), (short)0, (C), false, false)

// Load a 16-half fragment as two contiguous 8-half (16B) chunks, `gap` halves apart.
// A 16x32 f16, K order used here (same as B): lane<16 row=lane, K=kb..kb+15; lane>=16 row=lane-16,
//   K=kb+16..kb+31                                  -> gap = 8
// B-matrix 32x16 f16: lane<16 col=lane, K=kb..kb+15; lane>=16 K=kb+16..kb+31 -> gap = 8
__device__ __forceinline__ v16h load_frag(const _Float16* p, int gap) {
  v8h a = *(const v8h*)p;
  v8h b = *(const v8h*)(p + gap);
  v16h r;
#pragma unroll
  for (int i = 0; i < 8; ++i) { r[i] = a[i]; r[i + 8] = b[i]; }
  return r;
}

// ---------------------------------------------------------------------------
// 1) DFT twiddle matrix W[u][k] = exp(-2*pi*i*u*k/S), f16 row-major (symmetric).
// ---------------------------------------------------------------------------
__global__ __launch_bounds__(256) void k_twiddle(_Float16* __restrict__ wre,
                                                 _Float16* __restrict__ wim) {
  int t = blockIdx.x * 256 + threadIdx.x;      // 0 .. 16383
  int u = t >> 7, k = t & 127;
  int m = (u * k) & 127;                       // periodic: exact phase index
  float th = (2.0f * PIF / (float)S) * (float)m;
  wre[t] = (_Float16)cosf(th);
  wim[t] = (_Float16)(-sinf(th));
}

// ---------------------------------------------------------------------------
// 2) Transpose (B,S,S,C) f32 -> planar (B*C, S, S) f16.  One block per (b,i)
//    row-slab: async-stage 32 KB of f32 into LDS, then write 64 channel rows
//    as packed 16-byte stores.
// ---------------------------------------------------------------------------
__global__ __launch_bounds__(256) void k_transpose(const float* __restrict__ in,
                                                   _Float16* __restrict__ xh) {
  __shared__ float sF[S * CH];                 // 32 KB
  int bi = blockIdx.x;                         // b*128 + i
  int b = bi >> 7, i = bi & 127;
  const float* p = in + (size_t)bi * (S * CH);
#if HAVE_ASYNC_LDS
#pragma unroll
  for (int k = 0; k < 8; ++k) {
    int idx = (threadIdx.x + k * 256) * 4;     // 4 floats = 16 B per lane
    __builtin_amdgcn_global_load_async_to_lds_b128(
        (AS1 v4i*)(p + idx), (AS3 v4i*)(sF + idx), 0, 0);
  }
  async_wait0();
#else
#pragma unroll
  for (int k = 0; k < 8; ++k) {
    int idx = (threadIdx.x + k * 256) * 4;
    *(float4*)(sF + idx) = *(const float4*)(p + idx);
  }
#endif
  __syncthreads();
  int c  = threadIdx.x >> 2;                   // 64 channels, 4 threads each
  int j0 = (threadIdx.x & 3) << 5;             // 32 contiguous j per thread
  _Float16* q = xh + (((size_t)(b * CH + c)) << 14) + i * S + j0;
#pragma unroll
  for (int n8 = 0; n8 < 4; ++n8) {
    v8h pk;
#pragma unroll
    for (int n = 0; n < 8; ++n) pk[n] = (_Float16)sF[(j0 + n8 * 8 + n) * CH + c];
    *(v8h*)(q + n8 * 8) = pk;                  // 16-byte coalesced store
  }
}

// ---------------------------------------------------------------------------
// 3) Fused per-plane 2-D DFT via WMMA:  F = W * X * W  (W symmetric).
//    X async-staged into LDS (read 8x from LDS instead of 8x from global),
//    T^T kept in LDS, amp/phase stored TRANSPOSED so each lane writes 8
//    contiguous f16 (b128 stores). 96 KB dynamic LDS, 256 threads (8 waves).
// ---------------------------------------------------------------------------
__global__ __launch_bounds__(256) void k_dft_plane(
    const _Float16* __restrict__ xh, const _Float16* __restrict__ wre,
    const _Float16* __restrict__ wim, _Float16* __restrict__ ampH,
    _Float16* __restrict__ phH, unsigned int* __restrict__ chanMaxBits) {
  extern __shared__ _Float16 dynLds[];
  _Float16* sX   = dynLds;                     // 32 KB, X plane
  _Float16* sTre = dynLds + PELEMS;            // 32 KB, T^T real  [v][i]
  _Float16* sTim = dynLds + 2 * PELEMS;        // 32 KB, T^T imag
  int plane = blockIdx.x;
  int wave = threadIdx.x >> 5;
  int lane = threadIdx.x & 31;
  int aklo = (lane >> 4) << 4;                 // 0 | 16
  int bklo = (lane >> 4) << 4;                 // 0 | 16
  int lrow = lane & 15;
  const _Float16* Xp = xh + ((size_t)plane << 14);

  // ---- Stage 0: bulk-stage X plane (32 KB) into LDS ----
#if HAVE_ASYNC_LDS
#pragma unroll
  for (int k = 0; k < 8; ++k) {
    int idx = (threadIdx.x + k * 256) * 8;     // 8 halves = 16 B per lane
    __builtin_amdgcn_global_load_async_to_lds_b128(
        (AS1 v4i*)(Xp + idx), (AS3 v4i*)(sX + idx), 0, 0);
  }
  async_wait0();
#else
#pragma unroll
  for (int k = 0; k < 8; ++k) {
    int idx = (threadIdx.x + k * 256) * 8;
    *(v8h*)(sX + idx) = *(const v8h*)(Xp + idx);
  }
#endif
  __syncthreads();

  // ---- Stage 1: row DFT from LDS, result transposed into LDS ----
  for (int tile = wave; tile < 64; tile += 8) {
    int m0 = (tile >> 3) << 4, n0 = (tile & 7) << 4;
    int arow = m0 + lrow, bcol = n0 + lrow;
    v8f accRe = {}, accIm = {};
#pragma unroll
    for (int kb = 0; kb < S; kb += 32) {
      const _Float16* pw = wre + bcol * S + kb + bklo;
      __builtin_prefetch(pw + 64, 0, 1);       // global_prefetch_b8
      v16h a   = load_frag(sX + arow * S + kb + aklo, 8);
      v16h bre = load_frag(pw, 8);
      v16h bim = load_frag(wim + bcol * S + kb + bklo, 8);
      accRe = WMMA_F16(a, bre, accRe);
      accIm = WMMA_F16(a, bim, accIm);
    }
    int orow = m0 + ((lane >> 4) << 3);        // C/D layout: +8 rows for hi half
    int ocol = n0 + lrow;
#pragma unroll
    for (int r = 0; r < 8; ++r) {
      sTre[ocol * S + orow + r] = (_Float16)accRe[r];
      sTim[ocol * S + orow + r] = (_Float16)accIm[r];
    }
  }
  __syncthreads();

  // ---- Stage 2: column DFT from LDS; fuse amp/phase + plane max ----
  // Output stored transposed: amp[v][u] at linear v*S+u (packed b128 stores).
  float wmax = 0.0f;
  _Float16* ap = ampH + ((size_t)plane << 14);
  _Float16* pp = phH + ((size_t)plane << 14);
  for (int tile = wave; tile < 64; tile += 8) {
    int m0 = (tile >> 3) << 4, n0 = (tile & 7) << 4;
    int arow = m0 + lrow, bcol = n0 + lrow;
    v8f p1 = {}, p2 = {}, pim = {};            // Wre*Tre, Wim*Tim, Wre*Tim+Wim*Tre
#pragma unroll
    for (int kb = 0; kb < S; kb += 32) {
      v16h are = load_frag(wre + arow * S + kb + aklo, 8);
      v16h aim = load_frag(wim + arow * S + kb + aklo, 8);
      v16h bre = load_frag(&sTre[bcol * S + kb + bklo], 8);   // ds_load path
      v16h bim = load_frag(&sTim[bcol * S + kb + bklo], 8);
      p1  = WMMA_F16(are, bre, p1);
      p2  = WMMA_F16(aim, bim, p2);
      pim = WMMA_F16(are, bim, pim);
      pim = WMMA_F16(aim, bre, pim);
    }
    int orow = m0 + ((lane >> 4) << 3);        // = u base
    int ocol = n0 + lrow;                      // = v
    v8h pka, pkp;
#pragma unroll
    for (int r = 0; r < 8; ++r) {
      float re = p1[r] - p2[r];
      float im = pim[r];
      float amp = sqrtf(re * re + im * im);
      float ph  = atan2f(im, re) * (1.0f / PIF);
      pka[r] = (_Float16)amp;
      pkp[r] = (_Float16)ph;
      wmax = fmaxf(wmax, amp);
    }
    *(v8h*)(ap + ocol * S + orow) = pka;       // contiguous 16-byte store
    *(v8h*)(pp + ocol * S + orow) = pkp;
  }
  // wave32 max-reduce, then one atomic per wave (amp >= 0 -> uint-bit max exact)
#pragma unroll
  for (int o = 16; o > 0; o >>= 1) wmax = fmaxf(wmax, __shfl_xor(wmax, o, 32));
  if (lane == 0) atomicMax(&chanMaxBits[plane], __float_as_uint(wmax));
}

// ---------------------------------------------------------------------------
// 4) Per-pixel channel reduction (in transposed plane space; masks symmetric).
// ---------------------------------------------------------------------------
__global__ __launch_bounds__(256) void k_chan_reduce(
    const _Float16* __restrict__ ampH, const _Float16* __restrict__ phH,
    const unsigned int* __restrict__ chanMaxBits, float* __restrict__ phase_max,
    float* __restrict__ amp_max) {
  __shared__ float sInv[CH];
  int b = blockIdx.x >> 6;
  int pix = ((blockIdx.x & 63) << 8) + threadIdx.x;
  if (threadIdx.x < CH) {
    float m = __uint_as_float(chanMaxBits[b * CH + threadIdx.x]);
    sInv[threadIdx.x] = (m > 0.0f) ? 1.0f / m : 0.0f;
  }
  __syncthreads();
  float best = -1.0f;
  int bestC = 0;
  for (int c = 0; c < CH; ++c) {
    float a = (float)ampH[(((size_t)(b * CH + c)) << 14) + pix] * sInv[c];
    if (a > best) { best = a; bestC = c; }
  }
  float ph = (float)phH[(((size_t)(b * CH + bestC)) << 14) + pix];
  int i = pix >> 7, j = pix & 127;             // (v,u); upper mask is symmetric
  float upper = ((i + j) <= S) ? 1.0f : 0.0f;
  phase_max[b * PELEMS + pix] = ph;
  amp_max[b * PELEMS + pix] = best * upper;
}

// ---------------------------------------------------------------------------
// 5) 3x3 max-pool NMS + corner zero-frequency mask (both transpose-symmetric).
// ---------------------------------------------------------------------------
__global__ __launch_bounds__(256) void k_nms(const float* __restrict__ amp_max,
                                             float* __restrict__ scores) {
  int b = blockIdx.x >> 6;
  int pix = ((blockIdx.x & 63) << 8) + threadIdx.x;
  int i = pix >> 7, j = pix & 127;
  const float* A = amp_max + b * PELEMS;
  float c = A[pix], p = c;
#pragma unroll
  for (int di = -1; di <= 1; ++di)
#pragma unroll
    for (int dj = -1; dj <= 1; ++dj) {
      int ii = i + di, jj = j + dj;
      if (ii >= 0 && ii < S && jj >= 0 && jj < S) p = fmaxf(p, A[ii * S + jj]);
    }
  float nms = (c < p) ? 0.0f : c;
  bool zf = ((i < 2) || (i >= S - 2)) && ((j < 2) || (j >= S - 2));
  scores[b * PELEMS + pix] = zf ? 0.0f : nms;
}

// ---------------------------------------------------------------------------
// 6) Per-batch top-32.  Planes are stored transposed, so linear idx = v*S+u:
//    u = idx & 127, v = idx >> 7.
// ---------------------------------------------------------------------------
__global__ __launch_bounds__(256) void k_topk(
    float* __restrict__ scores, const float* __restrict__ phase_max,
    float* __restrict__ out, float* __restrict__ topP, float* __restrict__ topU,
    float* __restrict__ topV) {
  __shared__ float sv[256];
  __shared__ int si[256];
  __shared__ int sel[TOPK];
  int b = blockIdx.x, t = threadIdx.x;
  float* sc = scores + b * PELEMS;
  for (int k = 0; k < TOPK; ++k) {
    float bv = -2.0f;
    int bi = 0;
    for (int x = t; x < PELEMS; x += 256) {
      float v = sc[x];
      if (v > bv || (v == bv && x < bi)) { bv = v; bi = x; }
    }
    sv[t] = bv; si[t] = bi;
    __syncthreads();
    for (int o = 128; o > 0; o >>= 1) {
      if (t < o) {
        if (sv[t + o] > sv[t] || (sv[t + o] == sv[t] && si[t + o] < si[t])) {
          sv[t] = sv[t + o]; si[t] = si[t + o];
        }
      }
      __syncthreads();
    }
    if (t == 0) { sel[k] = si[0]; sc[si[0]] = -1.0f; }
    __syncthreads();
  }
  if (t < TOPK) {
    int idx = sel[t];
    int u = idx & 127, v = idx >> 7;           // transposed storage
    int su = (u > S / 2) ? u - S : u;
    int sv2 = (v > S / 2) ? v - S : v;
    float ph = phase_max[b * PELEMS + idx];
    out[b * TOPK + t] = ph;                    // top_phases (B,K,1)
    float* cd = out + BATCH * TOPK;            // coords (B,K,2)
    cd[(b * TOPK + t) * 2 + 0] = (float)su;
    cd[(b * TOPK + t) * 2 + 1] = (float)sv2;
    topP[b * TOPK + t] = ph;
    topU[b * TOPK + t] = (float)su;
    topV[b * TOPK + t] = (float)sv2;
  }
}

// ---------------------------------------------------------------------------
// 7) Cosine synthesis Iuvx[b,k,i,j] = cos(p*pi + (u*i + v*j)*2pi/S).
// ---------------------------------------------------------------------------
__global__ __launch_bounds__(256) void k_synth(const float* __restrict__ topP,
                                               const float* __restrict__ topU,
                                               const float* __restrict__ topV,
                                               float* __restrict__ out) {
  int plane = blockIdx.x;                      // b*TOPK + k
  float p = topP[plane] * PIF;
  float u = topU[plane], v = topV[plane];
  const float w = 2.0f * PIF / (float)S;
  float* o = out + (size_t)plane * PELEMS;
  for (int x = threadIdx.x; x < PELEMS; x += 256) {
    int i = x >> 7, j = x & 127;
    o[x] = cosf(p + w * (u * (float)i + v * (float)j));
  }
}

// ---------------------------------------------------------------------------
extern "C" void kernel_launch(void* const* d_in, const int* in_sizes, int n_in,
                              void* d_out, int out_size, void* d_ws, size_t ws_size,
                              hipStream_t stream) {
  (void)in_sizes; (void)n_in; (void)out_size; (void)ws_size;
  const float* in = (const float*)d_in[0];
  float* out = (float*)d_out;
  char* ws = (char*)d_ws;

  size_t off = 0;
  auto carve = [&](size_t bytes) {
    size_t p = off;
    off += (bytes + 255) & ~(size_t)255;
    return p;
  };
  _Float16* wre  = (_Float16*)(ws + carve(PELEMS * sizeof(_Float16)));
  _Float16* wim  = (_Float16*)(ws + carve(PELEMS * sizeof(_Float16)));
  _Float16* xh   = (_Float16*)(ws + carve((size_t)PLANES * PELEMS * sizeof(_Float16)));
  _Float16* ampH = (_Float16*)(ws + carve((size_t)PLANES * PELEMS * sizeof(_Float16)));
  _Float16* phH  = (_Float16*)(ws + carve((size_t)PLANES * PELEMS * sizeof(_Float16)));
  unsigned int* chanMax = (unsigned int*)(ws + carve(PLANES * sizeof(unsigned int)));
  float* phase_max = (float*)(ws + carve((size_t)BATCH * PELEMS * sizeof(float)));
  float* amp_max   = (float*)(ws + carve((size_t)BATCH * PELEMS * sizeof(float)));
  float* scores    = (float*)(ws + carve((size_t)BATCH * PELEMS * sizeof(float)));
  float* topP = (float*)(ws + carve(BATCH * TOPK * sizeof(float)));
  float* topU = (float*)(ws + carve(BATCH * TOPK * sizeof(float)));
  float* topV = (float*)(ws + carve(BATCH * TOPK * sizeof(float)));

  (void)hipMemsetAsync(chanMax, 0, PLANES * sizeof(unsigned int), stream);

  k_twiddle<<<PELEMS / 256, 256, 0, stream>>>(wre, wim);
  k_transpose<<<BATCH * S, 256, 0, stream>>>(in, xh);
  k_dft_plane<<<PLANES, 256, 3 * PELEMS * sizeof(_Float16), stream>>>(
      xh, wre, wim, ampH, phH, chanMax);
  k_chan_reduce<<<BATCH * 64, 256, 0, stream>>>(ampH, phH, chanMax, phase_max, amp_max);
  k_nms<<<BATCH * 64, 256, 0, stream>>>(amp_max, scores);
  k_topk<<<BATCH, 256, 0, stream>>>(scores, phase_max, out, topP, topU, topV);
  k_synth<<<BATCH * TOPK, 256, 0, stream>>>(topP, topU, topV,
                                            out + BATCH * TOPK + BATCH * TOPK * 2);
}
